// Backbone_4449586118738
// MI455X (gfx1250) — matrix operand order repaired and checked by reading, not compiled, not run on hardware
//
#include <hip/hip_runtime.h>

typedef __attribute__((ext_vector_type(16))) _Float16 v16h;
typedef __attribute__((ext_vector_type(8)))  _Float16 v8h;
typedef __attribute__((ext_vector_type(8)))  float    v8f;

#define EPS_ 1e-5f
#define B_   2048
#define L_   512
#define D_   7
#define M_   14336      /* B_*D_ */
#define DM_  48
#define PRED_ 96
#define H_   48
#define KLR  1536       /* 32*48 */

__device__ __forceinline__ float gelu_f(float x) {
  return 0.5f * x * (1.0f + erff(x * 0.70710678118654752f));
}

__device__ __forceinline__ float wsum32(float v) {
#pragma unroll
  for (int o = 16; o > 0; o >>= 1) v += __shfl_down(v, o, 32);
  return v;
}

// per-block partial (sum,sumsq) for 7 row-channels (d = row % 7)
__device__ __forceinline__ void write_stats7(const float* s, const float* q,
                                             float* partials, int nblk) {
  __shared__ float ls[8][14];
  int wave = threadIdx.x >> 5, lane = threadIdx.x & 31;
#pragma unroll
  for (int c = 0; c < 7; ++c) {
    float a = wsum32(s[c]);
    float b = wsum32(q[c]);
    if (lane == 0) { ls[wave][2 * c] = a; ls[wave][2 * c + 1] = b; }
  }
  __syncthreads();
  if (threadIdx.x < 14) {
    float t = 0.f;
#pragma unroll
    for (int w = 0; w < 8; ++w) t += ls[w][threadIdx.x];
    partials[(size_t)threadIdx.x * nblk + blockIdx.x] = t;
  }
}

// ---------------------------------------------------------------- constants
// B matrices are stored PRE-PACKED in WMMA B-fragment order:
//   chunk = (kTile * (N/16) + nTile) * 512 halves; within chunk lane*16 + h.
//   value = Bmat[kTile*32 + (lane<16?0:16) + h][nTile*16 + (lane&15)]
__device__ __forceinline__ void decode_pack(int j, int Ncols, int& krow, int& col) {
  int tileIdx = j >> 9;
  int lane = (j >> 4) & 31;
  int h = j & 15;
  int ntN = Ncols >> 4;
  int kbt = tileIdx / ntN, tn = tileIdx % ntN;
  krow = kbt * 32 + ((lane & 16) ? 16 : 0) + h;
  col = tn * 16 + (lane & 15);
}

__global__ void k_build_consts(_Float16* D2t, _Float16* DoS, float* c5,
                               _Float16* lrBt, _Float16* w1Bt,
                               const float* tf_fc_w, const float* linres_W,
                               const float* mlp_w1) {
  int i = blockIdx.x * blockDim.x + threadIdx.x;
  const double PI = 3.14159265358979323846;
  int krow, col;
  if (i < 262144) {                 // dct: Bmat[n][k] = 2*cos(pi*(n+.5)*k/L)
    decode_pack(i, 512, krow, col);
    D2t[i] = (_Float16)(float)(2.0 * cos(PI * (krow + 0.5) * col / 512.0));
  } else if (i < 524288) {          // idct: Bmat[kf][n] = S*s*cos(pi*(n+.5)*kf/L)
    int j = i - 262144;
    decode_pack(j, 512, krow, col);
    double coef = (krow == 0 ? 0.5 : 1.0) / 512.0;
    DoS[j] = (_Float16)(float)(coef * cos(PI * (col + 0.5) * krow / 512.0));
  } else if (i < 524800) {          // c5[l] = sum_k wf[k]*s[k]*cos(...)
    int l = i - 524288;
    double acc = 0.0;
    for (int k = 0; k < 5; ++k) {
      double sk = (k == 0) ? 0.044194173824159220 : 0.0625;
      acc += (double)tf_fc_w[k] * sk * cos(PI * (l + 0.5) * k / 512.0);
    }
    c5[l] = (float)acc;
  } else if (i < 672256) {          // linres_W^T packed  (K=1536, N=96)
    int j = i - 524800;
    decode_pack(j, 96, krow, col);
    lrBt[j] = (_Float16)linres_W[(size_t)col * KLR + krow];
  } else if (i < 696832) {          // mlp_w1^T packed (K=512, N=48)
    int j = i - 672256;
    decode_pack(j, 48, krow, col);
    w1Bt[j] = (_Float16)mlp_w1[(size_t)col * 512 + krow];
  }
}

// -------------------------------------------------------- transpose + f16
__global__ void k_transpose(const float* __restrict__ x, float* __restrict__ xp,
                            _Float16* __restrict__ Ah) {
  int tid = blockIdx.x * blockDim.x + threadIdx.x;
  if (tid >= B_ * L_ * D_) return;
  int b = tid / (L_ * D_);
  int r = tid - b * (L_ * D_);
  int l = r / D_, d = r - l * D_;
  float v = x[tid];
  size_t o = ((size_t)(b * D_ + d)) * L_ + l;
  xp[o] = v;
  Ah[o] = (_Float16)v;
}

// -------------------------------------------------------------- WMMA GEMM
// C[M,N] = A[M,K] @ Bpack (+bias[n]) (+res modes).
// One 16x16 M-tile x NT N-tiles per wave; B pre-packed in fragment order.
template <int NT>
__global__ void k_gemm_wmma(const _Float16* __restrict__ A,
                            const _Float16* __restrict__ Bp,
                            float* __restrict__ C, int M, int N, int K,
                            const float* __restrict__ bias,
                            const float* __restrict__ res, int res_mode,
                            const float* __restrict__ cw,
                            const float* __restrict__ cb) {
  int wavetile = blockIdx.x * (blockDim.x >> 5) + (threadIdx.x >> 5);
  int ntN = N >> 4;
  int ngrp = ntN / NT;
  int nwt = (M >> 4) * ngrp;
  if (wavetile >= nwt) return;
  int lane = threadIdx.x & 31;
  int tm = wavetile / ngrp;
  int tn0 = (wavetile % ngrp) * NT;
  int row = tm * 16 + (lane & 15);
  int koffA = (lane < 16) ? 0 : 16;
  size_t laneB = (size_t)lane * 16;

  v8f acc[NT];
#pragma unroll
  for (int t = 0; t < NT; ++t) acc[t] = {};

  const _Float16* Arow = A + (size_t)row * K;
  for (int kb = 0; kb < K; kb += 32) {
    __builtin_prefetch(Arow + kb + 64, 0, 3);
    const _Float16* Ap = Arow + kb + koffA;
    v8h alo = *(const v8h*)(Ap);
    v8h ahi = *(const v8h*)(Ap + 8);
    v16h a;
#pragma unroll
    for (int i = 0; i < 8; ++i) { a[i] = alo[i]; a[i + 8] = ahi[i]; }
    const _Float16* Bk = Bp + (((size_t)(kb >> 5) * ntN) << 9) + laneB;
#pragma unroll
    for (int t = 0; t < NT; ++t) {
      v16h b = *(const v16h*)(Bk + (((size_t)(tn0 + t)) << 9));
      acc[t] = __builtin_amdgcn_wmma_f32_16x16x32_f16(false, a, false, b,
                                                      (short)0, acc[t], false, false);
    }
  }
  int rbase = tm * 16 + ((lane < 16) ? 0 : 8);
#pragma unroll
  for (int t = 0; t < NT; ++t) {
    int col = (tn0 + t) * 16 + (lane & 15);
    float bi = bias ? bias[col] : 0.f;
#pragma unroll
    for (int v = 0; v < 8; ++v) {
      int r = rbase + v;
      float val = acc[t][v] + bi;
      if (res_mode == 1) {
        val += res[(size_t)r * N + col];
      } else if (res_mode == 2) {     // + dctconv(res): res*w[d]+b[d]
        int d = r % D_;
        val += res[(size_t)r * N + col] * cw[d] + cb[d];
      }
      C[(size_t)r * N + col] = val;
    }
  }
}

// ------------------------------------------------------------------ energy
__global__ void k_energy(const float* __restrict__ Z, float* __restrict__ energy) {
  int wave = threadIdx.x >> 5, lane = threadIdx.x & 31;
  int m = blockIdx.x * 8 + wave;
  const float* r = Z + (size_t)m * L_;
  float s = 0.f;
  for (int l = lane; l < L_; l += 32) { float v = r[l]; s += v * v; }
  s = wsum32(s);
  if (lane == 0) energy[m] = s;
}

__global__ void k_median_ne(const float* __restrict__ energy, float* __restrict__ ne) {
  int b = blockIdx.x * blockDim.x + threadIdx.x;
  if (b >= B_) return;
  float o[7], v[7];
#pragma unroll
  for (int i = 0; i < 7; ++i) { o[i] = energy[b * 7 + i]; v[i] = o[i]; }
#pragma unroll
  for (int i = 1; i < 7; ++i) {
    float key = v[i]; int j = i - 1;
    while (j >= 0 && v[j] > key) { v[j + 1] = v[j]; --j; }
    v[j + 1] = key;
  }
  float med = v[3] + 1e-6f;
#pragma unroll
  for (int i = 0; i < 7; ++i) ne[b * 7 + i] = o[i] / med;
}

// torch.quantile('linear') over 14336 values: single-block bitonic sort
__global__ void k_sort_quantile(const float* __restrict__ ne, float* sb,
                                const float* __restrict__ qp, float* thr) {
  const int NS = 16384;
  for (int i = threadIdx.x; i < NS; i += 1024) sb[i] = (i < M_) ? ne[i] : 3.0e38f;
  __syncthreads();
  for (int k = 2; k <= NS; k <<= 1) {
    for (int j = k >> 1; j > 0; j >>= 1) {
      for (int i = threadIdx.x; i < NS; i += 1024) {
        int ixj = i ^ j;
        if (ixj > i) {
          float a = sb[i], b = sb[ixj];
          bool up = ((i & k) == 0);
          if ((a > b) == up) { sb[i] = b; sb[ixj] = a; }
        }
      }
      __syncthreads();
    }
  }
  if (threadIdx.x == 0) {
    float q = qp[0];
    float pos = q * (float)(M_ - 1);
    int lo = (int)floorf(pos);
    if (lo < 0) lo = 0; if (lo > M_ - 1) lo = M_ - 1;
    int hi = lo + 1; if (hi > M_ - 1) hi = M_ - 1;
    float frac = pos - (float)lo;
    thr[0] = sb[lo] * (1.f - frac) + sb[hi] * frac;
  }
}

// u = gelu(dctconv(z_dct * mask)) in place, + stats per d
__global__ void k_fpass(float* __restrict__ u, const float* __restrict__ ne,
                        const float* __restrict__ thrp, const float* __restrict__ cw,
                        const float* __restrict__ cb, float* partials, int nblk) {
  float thr = thrp[0];
  float s[7] = {0}, q[7] = {0};
  size_t base = (size_t)blockIdx.x * 128 * L_;
  for (int e = threadIdx.x; e < 128 * L_; e += 256) {
    size_t idx = base + e;
    int m = (int)(idx >> 9);
    int d = m % D_;
    float msk = ne[m] > thr ? 1.f : 0.f;
    float v = u[idx] * msk * cw[d] + cb[d];
    v = gelu_f(v);
    u[idx] = v;
    s[d] += v; q[d] += v * v;
  }
  write_stats7(s, q, partials, nblk);
}

__global__ void k_finalize_stats(const float* __restrict__ partials, int nblk,
                                 int C, float invCnt, float* mv) {
  int c = threadIdx.x;
  if (c >= C) return;
  float s = 0.f, q = 0.f;
  for (int i = 0; i < nblk; ++i) {
    s += partials[(size_t)(2 * c) * nblk + i];
    q += partials[(size_t)(2 * c + 1) * nblk + i];
  }
  float m = s * invCnt;
  mv[2 * c] = m;
  mv[2 * c + 1] = q * invCnt - m * m;
}

__global__ void k_bn_to_f16(const float* __restrict__ u, const float* __restrict__ mv,
                            const float* __restrict__ g, const float* __restrict__ gb,
                            _Float16* __restrict__ Ah) {
  size_t idx = (size_t)blockIdx.x * blockDim.x + threadIdx.x;
  int d = (int)((idx >> 9) % D_);
  float v = (u[idx] - mv[2 * d]) * rsqrtf(mv[2 * d + 1] + EPS_) * g[d] + gb[d];
  Ah[idx] = (_Float16)v;
}

// ze[m][n*48+dm] = sum_p xp[m][n*16+p]*embed_W[dm][p] + embed_b[dm]
__global__ void k_patch_embed(const float* __restrict__ xp, const float* __restrict__ eW,
                              const float* __restrict__ eb, _Float16* __restrict__ ze) {
  size_t tid = (size_t)blockIdx.x * blockDim.x + threadIdx.x;
  if (tid >= (size_t)M_ * KLR) return;
  int m = (int)(tid / KLR);
  int r = (int)(tid % KLR);
  int n = r / DM_, dm = r % DM_;
  const float* xr = xp + (size_t)m * L_ + n * 16;
  const float* wr = eW + dm * 16;
  float acc = eb[dm];
#pragma unroll
  for (int p = 0; p < 16; ++p) acc += xr[p] * wr[p];
  ze[tid] = (_Float16)acc;
}

// depthwise conv k=3 stride=3 over ze -> z2[m][n*16+p], + stats per n
__global__ void k_depthwise(const _Float16* __restrict__ ze, const float* __restrict__ dw,
                            const float* __restrict__ db, float* __restrict__ z2,
                            float* partials, int nblk) {
  int c = threadIdx.x >> 3;       // channel n (0..31)
  int p0 = threadIdx.x & 7;
  float w0 = dw[c * 3], w1 = dw[c * 3 + 1], w2 = dw[c * 3 + 2], bb = db[c];
  float s = 0.f, q = 0.f;
  for (int r = 0; r < 32; ++r) {
    int m = blockIdx.x * 32 + r;
    const _Float16* zp = ze + (size_t)m * KLR + c * DM_;
    for (int pp = p0; pp < 16; pp += 8) {
      float v = (float)zp[pp * 3] * w0 + (float)zp[pp * 3 + 1] * w1 +
                (float)zp[pp * 3 + 2] * w2 + bb;
      z2[(size_t)m * L_ + c * 16 + pp] = v;
      s += v; q += v * v;
    }
  }
  __shared__ float ls[256][2];
  ls[threadIdx.x][0] = s; ls[threadIdx.x][1] = q;
  __syncthreads();
  if (threadIdx.x < 64) {
    int cc = threadIdx.x >> 1, isq = threadIdx.x & 1;
    float t = 0.f;
#pragma unroll
    for (int j = 0; j < 8; ++j) t += ls[cc * 8 + j][isq];
    partials[(size_t)threadIdx.x * nblk + blockIdx.x] = t;
  }
}

__global__ void k_depth_bn_gelu(float* __restrict__ z2, const float* __restrict__ mv,
                                const float* __restrict__ g, const float* __restrict__ gb) {
  size_t idx = (size_t)blockIdx.x * blockDim.x + threadIdx.x;
  int n = (int)((idx & 511) >> 4);
  float v = (z2[idx] - mv[2 * n]) * rsqrtf(mv[2 * n + 1] + EPS_) * g[n] + gb[n];
  z2[idx] = gelu_f(v);
}

// att[m] = dot((A[+B]), c5) + bf
__global__ void k_tf_dot(const float* __restrict__ A, const float* __restrict__ Bopt,
                         const float* __restrict__ c5, const float* __restrict__ bfp,
                         float* __restrict__ att) {
  int wave = threadIdx.x >> 5, lane = threadIdx.x & 31;
  int m = blockIdx.x * 8 + wave;
  const float* ra = A + (size_t)m * L_;
  const float* rb = Bopt ? Bopt + (size_t)m * L_ : nullptr;
  float acc = 0.f;
  for (int l = lane; l < L_; l += 32) {
    float v = ra[l];
    if (rb) v += rb[l];
    acc += v * c5[l];
  }
  acc = wsum32(acc);
  if (lane == 0) att[m] = acc + bfp[0];
}

__global__ void k_tf_stats(const float* __restrict__ att, float* mv) {
  int wave = threadIdx.x >> 5, lane = threadIdx.x & 31;
  float s = 0.f, q = 0.f;
  for (int i = threadIdx.x; i < M_; i += 1024) { float v = att[i]; s += v; q += v * v; }
  s = wsum32(s); q = wsum32(q);
  __shared__ float ls[32][2];
  if (lane == 0) { ls[wave][0] = s; ls[wave][1] = q; }
  __syncthreads();
  if (threadIdx.x == 0) {
    float S = 0.f, Q = 0.f;
    for (int w = 0; w < 32; ++w) { S += ls[w][0]; Q += ls[w][1]; }
    float mm = S / (float)M_;
    mv[0] = mm;
    mv[1] = Q / (float)M_ - mm * mm;
  }
}

__global__ void k_tf_apply(const float* __restrict__ att, const float* __restrict__ mv,
                           const float* __restrict__ tg, const float* __restrict__ tb,
                           const float* __restrict__ wc, const float* __restrict__ bc,
                           float* __restrict__ a) {
  int m = blockIdx.x * blockDim.x + threadIdx.x;
  if (m >= M_) return;
  float v = (att[m] - mv[0]) * rsqrtf(mv[1] + EPS_) * tg[0] + tb[0];
  v = gelu_f(v);
  float z = v * wc[0] + bc[0];
  a[m] = 1.f / (1.f + expf(-z));
}

// z1*=a; z2*=a; Z=z1+z2; stats(Z)
__global__ void k_scale_add(float* __restrict__ z1, float* __restrict__ z2,
                            const float* __restrict__ a1, float* __restrict__ Z,
                            float* partials, int nblk) {
  float s[7] = {0}, q[7] = {0};
  size_t base = (size_t)blockIdx.x * 128 * L_;
  for (int e = threadIdx.x; e < 128 * L_; e += 256) {
    size_t idx = base + e;
    int m = (int)(idx >> 9);
    int d = m % D_;
    float a = a1[m];
    float v1 = z1[idx] * a, v2 = z2[idx] * a;
    z1[idx] = v1; z2[idx] = v2;
    float zs = v1 + v2;
    Z[idx] = zs;
    s[d] += zs; q[d] += zs * zs;
  }
  write_stats7(s, q, partials, nblk);
}

// Z = gelu(bn(Z)) in place + stats of result
__global__ void k_bn_gelu_stats(float* __restrict__ Z, const float* __restrict__ mv,
                                const float* __restrict__ g, const float* __restrict__ gb,
                                float* partials, int nblk) {
  float s[7] = {0}, q[7] = {0};
  size_t base = (size_t)blockIdx.x * 128 * L_;
  for (int e = threadIdx.x; e < 128 * L_; e += 256) {
    size_t idx = base + e;
    int d = (int)((idx >> 9) % D_);
    float v = (Z[idx] - mv[2 * d]) * rsqrtf(mv[2 * d + 1] + EPS_) * g[d] + gb[d];
    v = gelu_f(v);
    Z[idx] = v;
    s[d] += v; q[d] += v * v;
  }
  write_stats7(s, q, partials, nblk);
}

// inter = gelu(bn(dctconv(zA)))*a2 ; z1'=z1*inter+z2 ; z2'=z2*inter+z1' ;
// Z = z1'*z2'+z1'+z2' ; stats(Z).  BN-of-affine derived: m'=w*m+b, v'=w^2*v.
__global__ void k_inter_combine(float* __restrict__ Z, const float* __restrict__ z1,
                                const float* __restrict__ z2, const float* __restrict__ a2,
                                const float* __restrict__ mv, const float* __restrict__ cw,
                                const float* __restrict__ cb, const float* __restrict__ g,
                                const float* __restrict__ gb, float* partials, int nblk) {
  float s[7] = {0}, q[7] = {0};
  size_t base = (size_t)blockIdx.x * 128 * L_;
  for (int e = threadIdx.x; e < 128 * L_; e += 256) {
    size_t idx = base + e;
    int m = (int)(idx >> 9);
    int d = m % D_;
    float w = cw[d];
    float zA = Z[idx];
    float inter = gelu_f((w * (zA - mv[2 * d])) *
                         rsqrtf(w * w * mv[2 * d + 1] + EPS_) * g[d] + gb[d]) * a2[m];
    float v1 = z1[idx] * inter + z2[idx];
    float v2 = z2[idx] * inter + v1;
    float zz = v1 * v2 + v1 + v2;
    Z[idx] = zz;
    s[d] += zz; q[d] += zz * zz;
  }
  write_stats7(s, q, partials, nblk);
}

__global__ void k_bn_gelu_to_f16(const float* __restrict__ Z, const float* __restrict__ mv,
                                 const float* __restrict__ g, const float* __restrict__ gb,
                                 _Float16* __restrict__ Ah) {
  size_t idx = (size_t)blockIdx.x * blockDim.x + threadIdx.x;
  int d = (int)((idx >> 9) % D_);
  float v = (Z[idx] - mv[2 * d]) * rsqrtf(mv[2 * d + 1] + EPS_) * g[d] + gb[d];
  Ah[idx] = (_Float16)gelu_f(v);
}

// h = gelu(h)*h in place + stats per d (rows of 48)
__global__ void k_mlp_gate(float* __restrict__ h, float* partials, int nblk) {
  float s[7] = {0}, q[7] = {0};
  size_t base = (size_t)blockIdx.x * 128 * H_;
  for (int e = threadIdx.x; e < 128 * H_; e += 256) {
    size_t idx = base + e;
    int m = (int)(idx / H_);
    int d = m % D_;
    float v = h[idx];
    v = gelu_f(v) * v;
    h[idx] = v;
    s[d] += v; q[d] += v * v;
  }
  write_stats7(s, q, partials, nblk);
}

// out[b][pred][d] = bn(hg) @ mlp_w2^T + mlp_b2 + z_res
__global__ void k_mlp_out(const float* __restrict__ h, const float* __restrict__ mv,
                          const float* __restrict__ mg, const float* __restrict__ mb,
                          const float* __restrict__ w2, const float* __restrict__ b2,
                          const float* __restrict__ zres, float* __restrict__ out) {
  int tid = blockIdx.x * blockDim.x + threadIdx.x;
  if (tid >= M_ * PRED_) return;
  int m = tid / PRED_, pd = tid % PRED_, d = m % D_;
  float mean = mv[2 * d], isd = rsqrtf(mv[2 * d + 1] + EPS_);
  float gg = mg[d], bb = mb[d];
  float acc = b2[pd] + zres[(size_t)m * PRED_ + pd];
  const float* hr = h + (size_t)m * H_;
  const float* wr = w2 + (size_t)pd * H_;
#pragma unroll
  for (int j = 0; j < H_; ++j) {
    float hn = (hr[j] - mean) * isd * gg + bb;
    acc += hn * wr[j];
  }
  out[(size_t)(m / D_) * (PRED_ * D_) + pd * D_ + (m % D_)] = acc;
}

// =========================================================================
extern "C" void kernel_launch(void* const* d_in, const int* in_sizes, int n_in,
                              void* d_out, int out_size, void* d_ws, size_t ws_size,
                              hipStream_t stream) {
  (void)in_sizes; (void)n_in; (void)out_size; (void)ws_size;
  const float* x          = (const float*)d_in[0];
  const float* dctconv_w  = (const float*)d_in[1];
  const float* dctconv_b  = (const float*)d_in[2];
  const float* dctnorm_g  = (const float*)d_in[3];
  const float* dctnorm_b  = (const float*)d_in[4];
  const float* threshold  = (const float*)d_in[5];
  const float* embed_W    = (const float*)d_in[6];
  const float* embed_b    = (const float*)d_in[7];
  const float* linres_W   = (const float*)d_in[8];
  const float* linres_b   = (const float*)d_in[9];
  const float* depth1_w   = (const float*)d_in[10];
  const float* depth1_b   = (const float*)d_in[11];
  const float* depthnorm_g= (const float*)d_in[12];
  const float* depthnorm_b= (const float*)d_in[13];
  const float* tf_fc_w    = (const float*)d_in[14];
  const float* tf_fc_b    = (const float*)d_in[15];
  const float* tf_norm_g  = (const float*)d_in[16];
  const float* tf_norm_b  = (const float*)d_in[17];
  const float* tf_conv_w  = (const float*)d_in[18];
  const float* tf_conv_b  = (const float*)d_in[19];
  const float* mlp_w1     = (const float*)d_in[20];
  const float* mlp_b1     = (const float*)d_in[21];
  const float* mlp_w2     = (const float*)d_in[22];
  const float* mlp_b2     = (const float*)d_in[23];
  const float* mlpnorm_g  = (const float*)d_in[24];
  const float* mlpnorm_b  = (const float*)d_in[25];
  float* out = (float*)d_out;

  char* wsp = (char*)d_ws;
  size_t off = 0;
  auto alloc = [&](size_t bytes) -> char* {
    char* p = wsp + off;
    off = (off + bytes + 255) & ~((size_t)255);
    return p;
  };
  const size_t ML = (size_t)M_ * L_;
  float*     xp    = (float*)alloc(ML * 4);
  _Float16*  Ah    = (_Float16*)alloc(ML * 2);      // reused: xp_h, z_h, zfin_h
  float*     bufZ  = (float*)alloc(ML * 4);          // z_dct, then Z
  float*     z1    = (float*)alloc(ML * 4);
  float*     z2    = (float*)alloc(ML * 4);
  _Float16*  D2t   = (_Float16*)alloc(512 * 512 * 2);
  _Float16*  DoS   = (_Float16*)alloc(512 * 512 * 2);
  float*     c5    = (float*)alloc(512 * 4);
  _Float16*  ze    = (_Float16*)alloc((size_t)M_ * KLR * 2);
  _Float16*  lrBt  = (_Float16*)alloc((size_t)KLR * 96 * 2);
  _Float16*  w1Bt  = (_Float16*)alloc(512 * 48 * 2);
  float*     zres  = (float*)alloc((size_t)M_ * PRED_ * 4);
  float*     hbuf  = (float*)alloc((size_t)M_ * H_ * 4);
  float*     energy= (float*)alloc(M_ * 4);
  float*     ne    = (float*)alloc(M_ * 4);
  float*     att   = (float*)alloc(M_ * 4);
  float*     a1    = (float*)alloc(M_ * 4);
  float*     a2    = (float*)alloc(M_ * 4);
  float*     sortb = (float*)alloc(16384 * 4);
  float*     part  = (float*)alloc(64 * 512 * 4);
  float*     mv    = (float*)alloc(128 * 4);         // [0..13] ch7, [32..33] tf
  float*     mvd   = (float*)alloc(64 * 4);          // depth 32-ch stats
  float*     thr   = (float*)alloc(4);

  const float invBL = 1.f / (float)((size_t)B_ * L_);

  // constants + transpose
  k_build_consts<<<2722, 256, 0, stream>>>(D2t, DoS, c5, lrBt, w1Bt,
                                           tf_fc_w, linres_W, mlp_w1);
  k_transpose<<<28672, 256, 0, stream>>>(x, xp, Ah);

  // GEMM1: z_dct = xp @ D2^T   (896 M-tiles x 8 groups of 4 N-tiles)
  k_gemm_wmma<4><<<896, 256, 0, stream>>>(Ah, D2t, bufZ, M_, 512, 512,
                                          nullptr, nullptr, 0, nullptr, nullptr);
  // adaptive mask
  k_energy<<<1792, 256, 0, stream>>>(bufZ, energy);
  k_median_ne<<<8, 256, 0, stream>>>(energy, ne);
  k_sort_quantile<<<1, 1024, 0, stream>>>(ne, sortb, threshold, thr);
  // gelu(dctconv(z_dct*mask)) + BN stats, then BN -> f16
  k_fpass<<<112, 256, 0, stream>>>(bufZ, ne, thr, dctconv_w, dctconv_b, part, 112);
  k_finalize_stats<<<1, 32, 0, stream>>>(part, 112, 7, invBL, mv);
  k_bn_to_f16<<<28672, 256, 0, stream>>>(bufZ, mv, dctnorm_g, dctnorm_b, Ah);
  // GEMM2: z1 = zh @ (S*Do) + dctconv(xp)
  k_gemm_wmma<4><<<896, 256, 0, stream>>>(Ah, DoS, z1, M_, 512, 512,
                                          nullptr, xp, 2, dctconv_w, dctconv_b);
  // patching branch
  k_patch_embed<<<86016, 256, 0, stream>>>(xp, embed_W, embed_b, ze);
  // GEMM3: z_res = ze @ linres_W^T + linres_b   (896 x 2 groups of 3)
  k_gemm_wmma<3><<<224, 256, 0, stream>>>(ze, lrBt, zres, M_, 96, KLR,
                                          linres_b, nullptr, 0, nullptr, nullptr);
  k_depthwise<<<448, 256, 0, stream>>>(ze, depth1_w, depth1_b, z2, part, 448);
  k_finalize_stats<<<1, 32, 0, stream>>>(part, 448, 32, 1.f / 229376.f, mvd);
  k_depth_bn_gelu<<<28672, 256, 0, stream>>>(z2, mvd, depthnorm_g, depthnorm_b);
  // gating 1: att = tf(z1+z2)
  k_tf_dot<<<1792, 256, 0, stream>>>(z1, z2, c5, tf_fc_b, att);
  k_tf_stats<<<1, 1024, 0, stream>>>(att, mv + 32);
  k_tf_apply<<<56, 256, 0, stream>>>(att, mv + 32, tf_norm_g, tf_norm_b,
                                     tf_conv_w, tf_conv_b, a1);
  k_scale_add<<<112, 256, 0, stream>>>(z1, z2, a1, bufZ, part, 112);
  k_finalize_stats<<<1, 32, 0, stream>>>(part, 112, 7, invBL, mv);
  // gating 2: z_att = tf(Z)
  k_tf_dot<<<1792, 256, 0, stream>>>(bufZ, nullptr, c5, tf_fc_b, att);
  k_tf_stats<<<1, 1024, 0, stream>>>(att, mv + 32);
  k_tf_apply<<<56, 256, 0, stream>>>(att, mv + 32, tf_norm_g, tf_norm_b,
                                     tf_conv_w, tf_conv_b, a2);
  // zA = gelu(bn(Z)); inter/combine; final bn+gelu
  k_bn_gelu_stats<<<112, 256, 0, stream>>>(bufZ, mv, dctnorm_g, dctnorm_b, part, 112);
  k_finalize_stats<<<1, 32, 0, stream>>>(part, 112, 7, invBL, mv);
  k_inter_combine<<<112, 256, 0, stream>>>(bufZ, z1, z2, a2, mv, dctconv_w,
                                           dctconv_b, dctnorm_g, dctnorm_b, part, 112);
  k_finalize_stats<<<1, 32, 0, stream>>>(part, 112, 7, invBL, mv);
  k_bn_gelu_to_f16<<<28672, 256, 0, stream>>>(bufZ, mv, dctnorm_g, dctnorm_b, Ah);
  // GEMM4: h = zfin @ mlp_w1^T + mlp_b1   (896 x 1 group of 3)
  k_gemm_wmma<3><<<112, 256, 0, stream>>>(Ah, w1Bt, hbuf, M_, 48, 512,
                                          mlp_b1, nullptr, 0, nullptr, nullptr);
  k_mlp_gate<<<112, 256, 0, stream>>>(hbuf, part, 112);
  k_finalize_stats<<<1, 32, 0, stream>>>(part, 112, 7, 1.f / 98304.f, mv);
  k_mlp_out<<<5376, 256, 0, stream>>>(hbuf, mv, mlpnorm_g, mlpnorm_b,
                                      mlp_w2, mlp_b2, zres, out);
}
